// TransMIL_72791105732985
// MI455X (gfx1250) — 16-bit and 8-bit matrix operand order repaired and checked by reading, compiled, not run on hardware
//
#include <hip/hip_runtime.h>

typedef __bf16 bf16;
typedef __bf16 v16bf __attribute__((ext_vector_type(16)));
typedef float  v8f   __attribute__((ext_vector_type(8)));
typedef int    v16i  __attribute__((ext_vector_type(16)));

#define NP     20992
#define ND     20737
#define DM     512
#define NHEADS 8
#define HD     64
#define LM     256
#define LGRP   82
#define NPOOL  20480
#define EMB    1024
#define HW     144

union FragU { v16bf v; uint4 q[2]; };
union Frag8 { v16i v; uint2 d2[8]; uint4 d4[4]; };

__device__ __forceinline__ unsigned char f32_to_e4m3(float x) {
  unsigned u = __float_as_uint(x);
  unsigned sign = (u >> 24) & 0x80u;
  float ax = __uint_as_float(u & 0x7FFFFFFFu);
  if (!(ax == ax)) return (unsigned char)(sign | 0x7Fu);
  if (ax >= 464.f) return (unsigned char)(sign | 0x7Eu);
  if (ax < 0.015625f) {
    int q = (int)(ax * 512.f + 0.5f);
    return (unsigned char)(sign | (q & 0x7));
  }
  int e = (int)((u >> 23) & 0xFF) - 127;
  float scale = exp2f((float)(3 - e));
  int q = (int)(ax * scale + 0.5f);
  if (q >= 16) { q >>= 1; e += 1; }
  int E = e + 7;
  return (unsigned char)(sign | (E << 3) | (q & 0x7));
}

__device__ __forceinline__ void ld_a(FragU& f, const bf16* aBase, int k0) {
  f.q[0] = *(const uint4*)(aBase + k0);
  f.q[1] = *(const uint4*)(aBase + k0 + 16);
}
__device__ __forceinline__ void ld_b(FragU* f, const bf16* BT, int bcol, int ldbt, int k0, int kb) {
#pragma unroll
  for (int t = 0; t < 4; ++t) {
    const bf16* bp = BT + (long)(bcol + 16 * t) * ldbt + k0 + kb;
    f[t].q[0] = *(const uint4*)(bp);
    f[t].q[1] = *(const uint4*)(bp + 16);
  }
}

__global__ __launch_bounds__(256) void gemm_bf16_wmma(
    const bf16* __restrict__ A, const bf16* __restrict__ BT, float* __restrict__ C,
    int K, int lda, int ldbt, int ldc,
    long sA, long sB, long sC,
    const float* __restrict__ bias, int mode, int rowOff)
{
  int b = blockIdx.z;
  A  += (long)b * sA;
  BT += (long)b * sB;
  C  += (long)b * sC;
  int wave = threadIdx.x >> 5;
  int lane = threadIdx.x & 31;
  int m0 = blockIdx.y * 128 + wave * 16;
  int n0 = blockIdx.x * 64;
  int arow = m0 + (lane & 15);
  int klo  = (lane < 16) ? 0 : 8;
  int kb   = (lane < 16) ? 0 : 8;
  int bcol = n0 + (lane & 15);
  const bf16* aBase = A + (long)arow * lda + klo;
  v8f acc[4] = {};
  FragU afA, afB, bfA[4], bfB[4];
  ld_a(afA, aBase, 0);  ld_b(bfA, BT, bcol, ldbt, 0, kb);
  ld_a(afB, aBase, 32); ld_b(bfB, BT, bcol, ldbt, 32, kb);
  for (int k0 = 0; k0 < K; k0 += 64) {
    __builtin_prefetch(aBase + k0 + 128, 0, 0);
#pragma unroll
    for (int t = 0; t < 4; ++t)
      acc[t] = __builtin_amdgcn_wmma_f32_16x16x32_bf16(
          false, afA.v, false, bfA[t].v, (short)0, acc[t], false, false);
    if (k0 + 64 < K) { ld_a(afA, aBase, k0 + 64); ld_b(bfA, BT, bcol, ldbt, k0 + 64, kb); }
#pragma unroll
    for (int t = 0; t < 4; ++t)
      acc[t] = __builtin_amdgcn_wmma_f32_16x16x32_bf16(
          false, afB.v, false, bfB[t].v, (short)0, acc[t], false, false);
    if (k0 + 96 < K) { ld_a(afB, aBase, k0 + 96); ld_b(bfB, BT, bcol, ldbt, k0 + 96, kb); }
  }
  int rbase = m0 + ((lane >> 4) << 3);
  int cbase = n0 + (lane & 15);
#pragma unroll
  for (int t = 0; t < 4; ++t) {
    int cc = cbase + 16 * t;
#pragma unroll
    for (int r = 0; r < 8; ++r) {
      int row = rbase + r;
      float vv = acc[t][r];
      if (mode == 0) {
        C[(long)row * ldc + cc] = vv;
      } else if (mode == 1) {
        vv += bias[cc];
        C[(long)row * ldc + cc] = vv > 0.f ? vv : 0.f;
      } else {
        if (row >= rowOff) C[(long)(row - rowOff) * ldc + cc] += vv + bias[cc];
      }
    }
  }
}

__global__ __launch_bounds__(256) void gemm_fp8_wmma(
    const unsigned char* __restrict__ A, const unsigned char* __restrict__ BT,
    float* __restrict__ C, int K, int lda, int ldbt, int ldc, long sA, long sB, long sC)
{
  int b = blockIdx.z;
  A  += (long)b * sA;
  BT += (long)b * sB;
  C  += (long)b * sC;
  int wave = threadIdx.x >> 5;
  int lane = threadIdx.x & 31;
  int m0 = blockIdx.y * 128 + wave * 16;
  int n0 = blockIdx.x * 64;
  int arow = m0 + (lane & 15);
  int klo8 = (lane < 16) ? 0 : 16;
  int kb8  = (lane < 16) ? 0 : 16;
  const unsigned char* aBase = A + (long)arow * lda;
  v8f acc[4] = {};
  for (int k0 = 0; k0 < K; k0 += 128) {
    Frag8 af;
#pragma unroll
    for (int hlf = 0; hlf < 2; ++hlf)
#pragma unroll
      for (int j = 0; j < 4; ++j)
        af.d2[hlf * 4 + j] = *(const uint2*)(aBase + k0 + hlf * 64 + klo8 + 32 * (j >> 1) + 8 * (j & 1));
    __builtin_prefetch(aBase + k0 + 256, 0, 0);
#pragma unroll
    for (int t = 0; t < 4; ++t) {
      Frag8 bfr;
      const unsigned char* bp = BT + (long)(n0 + 16 * t + (lane & 15)) * ldbt + k0 + kb8;
#pragma unroll
      for (int g = 0; g < 4; ++g) bfr.d4[g] = *(const uint4*)(bp + 32 * g);
      acc[t] = __builtin_amdgcn_wmma_f32_16x16x128_fp8_fp8(
          af.v, bfr.v, (short)0, acc[t], false, false);
    }
  }
  int rbase = m0 + ((lane >> 4) << 3);
  int cbase = n0 + (lane & 15);
#pragma unroll
  for (int t = 0; t < 4; ++t) {
    int cc = cbase + 16 * t;
#pragma unroll
    for (int r = 0; r < 8; ++r)
      C[(long)(rbase + r) * ldc + cc] = acc[t][r];
  }
}

__global__ void pool_bf16_kernel(const float* __restrict__ x, bf16* __restrict__ o)
{
  long t = (long)blockIdx.x * 256 + threadIdx.x;
  if (t >= (long)NPOOL * EMB) return;
  long e = t & (EMB - 1); long r = t >> 10;
  float a = x[(2 * r) * EMB + e], b = x[(2 * r + 1) * EMB + e];
  o[t] = (bf16)fmaxf(a, b);
}

__global__ void transpose_bf16_kernel(const float* __restrict__ W, bf16* __restrict__ WT, int K, int N)
{
  long t = (long)blockIdx.x * 256 + threadIdx.x;
  if (t >= (long)K * N) return;
  int n = (int)(t % N); int k = (int)(t / N);
  WT[(long)n * K + k] = (bf16)W[t];
}

__global__ void build_h_kernel(const float* __restrict__ ft, const float* __restrict__ cls,
                               float* __restrict__ h)
{
  long t = (long)blockIdx.x * 256 + threadIdx.x;
  if (t >= (long)ND * DM) return;
  int c = (int)(t & (DM - 1)); long r = t >> 9;
  float v;
  if (r == 0)          v = cls[c];
  else if (r <= NPOOL) v = ft[(r - 1) * DM + c];
  else                 v = ft[(r - 1 - NPOOL) * DM + c];
  h[t] = v;
}

__global__ __launch_bounds__(256) void ln_pad_kernel(const float* __restrict__ h,
    const float* __restrict__ w, const float* __restrict__ b, bf16* __restrict__ o)
{
  int rp = blockIdx.x;
  bf16* orow = o + (long)rp * DM;
  if (rp < 255) {
    for (int c = threadIdx.x; c < DM; c += 256) orow[c] = (bf16)0.f;
    return;
  }
  const float* row = h + (long)(rp - 255) * DM;
  __shared__ float red[256];
  int t = threadIdx.x;
  float s = 0.f;
  for (int c = t; c < DM; c += 256) s += row[c];
  red[t] = s; __syncthreads();
  for (int st = 128; st > 0; st >>= 1) { if (t < st) red[t] += red[t + st]; __syncthreads(); }
  float mu = red[0] / DM; __syncthreads();
  float vs = 0.f;
  for (int c = t; c < DM; c += 256) { float d = row[c] - mu; vs += d * d; }
  red[t] = vs; __syncthreads();
  for (int st = 128; st > 0; st >>= 1) { if (t < st) red[t] += red[t + st]; __syncthreads(); }
  float inv = rsqrtf(red[0] / DM + 1e-5f);
  for (int c = t; c < DM; c += 256) orow[c] = (bf16)((row[c] - mu) * inv * w[c] + b[c]);
}

__global__ void split_qkv_kernel(const float* __restrict__ qkvf,
    bf16* __restrict__ qb, bf16* __restrict__ kb, bf16* __restrict__ vb)
{
  long t = (long)blockIdx.x * 256 + threadIdx.x;
  if (t >= (long)NP * DM) return;
  int c = (int)(t & (DM - 1)); long i = t >> 9;
  int hh = c >> 6; int d = c & 63;
  const float* row = qkvf + i * (3 * DM);
  long o = (long)hh * NP * HD + i * HD + d;
  qb[o] = (bf16)(0.125f * row[c]);
  kb[o] = (bf16)(row[DM + c]);
  vb[o] = (bf16)(row[2 * DM + c]);
}

__global__ void landmarks_kernel(const float* __restrict__ qkvf,
    float* __restrict__ qlf, float* __restrict__ klf,
    bf16* __restrict__ qlb, bf16* __restrict__ klb)
{
  int t = blockIdx.x * 256 + threadIdx.x;
  int d = t & 63; int m = (t >> 6) & 255; int hh = t >> 14;
  const float* base = qkvf + (long)(m * LGRP) * (3 * DM) + hh * HD + d;
  float sq = 0.f, sk = 0.f;
  for (int j = 0; j < LGRP; ++j) {
    sq += base[(long)j * (3 * DM)];
    sk += base[(long)j * (3 * DM) + DM];
  }
  float q = sq * (0.125f / (float)LGRP);
  float k = sk * (1.f / (float)LGRP);
  qlf[t] = q; klf[t] = k;
  qlb[t] = (bf16)q; klb[t] = (bf16)k;
}

__global__ __launch_bounds__(256) void a2_softmax_kernel(const float* __restrict__ ql,
    const float* __restrict__ kl, float* __restrict__ a2)
{
  int hh = blockIdx.y, m = blockIdx.x, c = threadIdx.x;
  const float* q  = ql + ((long)hh * LM + m) * HD;
  const float* kk = kl + ((long)hh * LM + c) * HD;
  float d = 0.f;
#pragma unroll 8
  for (int i = 0; i < HD; ++i) d += q[i] * kk[i];
  __shared__ float red[256];
  red[c] = d; __syncthreads();
  for (int st = 128; st > 0; st >>= 1) { if (c < st) red[c] = fmaxf(red[c], red[c + st]); __syncthreads(); }
  float mx = red[0]; __syncthreads();
  float e = __expf(d - mx);
  red[c] = e; __syncthreads();
  for (int st = 128; st > 0; st >>= 1) { if (c < st) red[c] += red[c + st]; __syncthreads(); }
  a2[(long)hh * LM * LM + (long)m * LM + c] = e / red[0];
}

__global__ __launch_bounds__(256) void pinv_init_kernel(const float* __restrict__ a2,
                                                        float* __restrict__ z)
{
  int hh = blockIdx.x;
  const float* x = a2 + (long)hh * LM * LM;
  int t = threadIdx.x;
  float rs = 0.f, cs = 0.f;
  for (int j = 0; j < LM; ++j) { rs += fabsf(x[t * LM + j]); cs += fabsf(x[j * LM + t]); }
  __shared__ float r1[256], r2[256];
  r1[t] = rs; r2[t] = cs; __syncthreads();
  for (int st = 128; st > 0; st >>= 1) {
    if (t < st) { r1[t] = fmaxf(r1[t], r1[t + st]); r2[t] = fmaxf(r2[t], r2[t + st]); }
    __syncthreads();
  }
  float inv = 1.f / (r1[0] * r2[0]);
  float* zr = z + (long)hh * LM * LM + (long)t * LM;
  for (int j = 0; j < LM; ++j) zr[j] = x[j * LM + t] * inv;
}

__global__ __launch_bounds__(256) void mm256_kernel(const float* __restrict__ A,
    const float* __restrict__ B, float* __restrict__ C,
    float alpha, float scale, int useI, float bI, int useBI)
{
  int bz = blockIdx.z;
  A += (long)bz * LM * LM; B += (long)bz * LM * LM; C += (long)bz * LM * LM;
  __shared__ float As[16][16];
  __shared__ float Bs[16][17];
  int tx = threadIdx.x, ty = threadIdx.y;
  int row = blockIdx.y * 16 + ty, col = blockIdx.x * 16 + tx;
  float acc = 0.f;
  for (int k0 = 0; k0 < LM; k0 += 16) {
    As[ty][tx] = A[row * LM + k0 + tx];
    float bv = B[(k0 + ty) * LM + col];
    Bs[ty][tx] = useBI ? (bI * ((k0 + ty) == col ? 1.f : 0.f) - bv) : bv;
    __syncthreads();
#pragma unroll
    for (int kk = 0; kk < 16; ++kk) acc += As[ty][kk] * Bs[kk][tx];
    __syncthreads();
  }
  C[row * LM + col] = useI ? (alpha * (row == col ? 1.f : 0.f) - acc) : scale * acc;
}

__global__ __launch_bounds__(256) void softmax_rows_kernel(const float* __restrict__ S,
    bf16* __restrict__ O, int C, long rsIn, long rsOut, long bsIn, long bsOut)
{
  const float* s = S + (long)blockIdx.y * bsIn + (long)blockIdx.x * rsIn;
  bf16* o = O + (long)blockIdx.y * bsOut + (long)blockIdx.x * rsOut;
  __shared__ float red[256];
  int t = threadIdx.x;
  float mx = -3.4e38f;
  for (int c = t; c < C; c += 256) mx = fmaxf(mx, s[c]);
  red[t] = mx; __syncthreads();
  for (int st = 128; st > 0; st >>= 1) { if (t < st) red[t] = fmaxf(red[t], red[t + st]); __syncthreads(); }
  mx = red[0]; __syncthreads();
  float sum = 0.f;
  for (int c = t; c < C; c += 256) sum += __expf(s[c] - mx);
  red[t] = sum; __syncthreads();
  for (int st = 128; st > 0; st >>= 1) { if (t < st) red[t] += red[t + st]; __syncthreads(); }
  float inv = 1.f / red[0];
  for (int c = t; c < C; c += 256) o[c] = (bf16)(__expf(s[c] - mx) * inv);
}

__global__ __launch_bounds__(256) void softmax_rows_fp8_kernel(const float* __restrict__ S,
    unsigned char* __restrict__ O, int C, long rsIn, long rsOut, long bsIn, long bsOut)
{
  const float* s = S + (long)blockIdx.y * bsIn + (long)blockIdx.x * rsIn;
  unsigned char* o = O + (long)blockIdx.y * bsOut + (long)blockIdx.x * rsOut;
  __shared__ float red[256];
  int t = threadIdx.x;
  float mx = -3.4e38f;
  for (int c = t; c < C; c += 256) mx = fmaxf(mx, s[c]);
  red[t] = mx; __syncthreads();
  for (int st = 128; st > 0; st >>= 1) { if (t < st) red[t] = fmaxf(red[t], red[t + st]); __syncthreads(); }
  mx = red[0]; __syncthreads();
  float sum = 0.f;
  for (int c = t; c < C; c += 256) sum += __expf(s[c] - mx);
  red[t] = sum; __syncthreads();
  for (int st = 128; st > 0; st >>= 1) { if (t < st) red[t] += red[t + st]; __syncthreads(); }
  float inv = 1.f / red[0];
  for (int c = t; c < C; c += 256) o[c] = f32_to_e4m3(__expf(s[c] - mx) * inv);
}

__global__ __launch_bounds__(256) void a3v_kernel(const bf16* __restrict__ a3b,
    const bf16* __restrict__ vb, float* __restrict__ av)
{
  int hh = blockIdx.y;
  int m = blockIdx.x * 4 + (threadIdx.x >> 6);
  int d = threadIdx.x & 63;
  const bf16* a = a3b + (long)hh * LM * NP + (long)m * NP;
  const bf16* v = vb + (long)hh * NP * HD + d;
  float acc = 0.f;
  for (int k = 0; k < NP; ++k) acc += (float)a[k] * (float)v[(long)k * HD];
  av[hh * LM * HD + m * HD + d] = acc;
}

__global__ void zc_kernel(const float* __restrict__ z, const float* __restrict__ av,
                          unsigned char* __restrict__ zcT)
{
  int t = blockIdx.x * 256 + threadIdx.x;
  int d = t & 63; int m = (t >> 6) & 255; int hh = t >> 14;
  const float* zr = z + (long)hh * LM * LM + (long)m * LM;
  const float* a = av + hh * LM * HD + d;
  float acc = 0.f;
  for (int k = 0; k < LM; ++k) acc += zr[k] * a[k * HD];
  zcT[(long)hh * HD * LM + (long)d * LM + m] = f32_to_e4m3(acc);
}

__global__ void res_add_kernel(const float* __restrict__ qkvf, const float* __restrict__ rw,
                               float* __restrict__ outf)
{
  long t = (long)blockIdx.x * 256 + threadIdx.x;
  if (t >= (long)NP * DM) return;
  int c = (int)(t & (DM - 1)); long i = t >> 9;
  int hh = c >> 6;
  float acc = 0.f;
  for (int j = 0; j < 33; ++j) {
    long r = i - 16 + j;
    if (r >= 0 && r < NP) acc += qkvf[r * (3 * DM) + 2 * DM + c] * rw[hh * 33 + j];
  }
  outf[t] += acc;
}

__global__ void cvt_bf16_kernel(const float* __restrict__ in, bf16* __restrict__ out, long n)
{
  long t = (long)blockIdx.x * 256 + threadIdx.x;
  if (t < n) out[t] = (bf16)in[t];
}

__global__ void ppeg_kernel(const float* __restrict__ hin, float* __restrict__ hout,
    const float* __restrict__ w7, const float* __restrict__ b7,
    const float* __restrict__ w5, const float* __restrict__ b5,
    const float* __restrict__ w3, const float* __restrict__ b3)
{
  long t = (long)blockIdx.x * 256 + threadIdx.x;
  if (t >= (long)ND * DM) return;
  int c = (int)(t & (DM - 1)); long r = t >> 9;
  if (r == 0) { hout[t] = hin[t]; return; }
  long p = r - 1; int hh = (int)(p / HW), ww = (int)(p % HW);
  const float* f = hin + DM;
  float acc = f[p * DM + c] + b7[c] + b5[c] + b3[c];
  for (int dy = -3; dy <= 3; ++dy) {
    int y = hh + dy; if (y < 0 || y >= HW) continue;
    for (int dx = -3; dx <= 3; ++dx) {
      int xx = ww + dx; if (xx < 0 || xx >= HW) continue;
      float vv = f[((long)y * HW + xx) * DM + c];
      acc += vv * w7[c * 49 + (dy + 3) * 7 + (dx + 3)];
      if (dy >= -2 && dy <= 2 && dx >= -2 && dx <= 2)
        acc += vv * w5[c * 25 + (dy + 2) * 5 + (dx + 2)];
      if (dy >= -1 && dy <= 1 && dx >= -1 && dx <= 1)
        acc += vv * w3[c * 9 + (dy + 1) * 3 + (dx + 1)];
    }
  }
  hout[t] = acc;
}

__global__ __launch_bounds__(512) void final_kernel(const float* __restrict__ h,
    const float* __restrict__ nw, const float* __restrict__ nb,
    const float* __restrict__ fw, const float* __restrict__ fb, float* __restrict__ out)
{
  __shared__ float red[512];
  __shared__ float xn[512];
  __shared__ float logits[4];
  int t = threadIdx.x;
  float v = h[t];
  red[t] = v; __syncthreads();
  for (int st = 256; st > 0; st >>= 1) { if (t < st) red[t] += red[t + st]; __syncthreads(); }
  float mu = red[0] / 512.f; __syncthreads();
  float d = v - mu;
  red[t] = d * d; __syncthreads();
  for (int st = 256; st > 0; st >>= 1) { if (t < st) red[t] += red[t + st]; __syncthreads(); }
  float inv = rsqrtf(red[0] / 512.f + 1e-5f);
  xn[t] = d * inv * nw[t] + nb[t];
  __syncthreads();
  if (t < 4) {
    float a = 0.f;
    for (int i = 0; i < 512; ++i) a += xn[i] * fw[i * 4 + t];
    logits[t] = a + fb[t];
  }
  __syncthreads();
  if (t == 0) {
    float mx = fmaxf(fmaxf(logits[0], logits[1]), fmaxf(logits[2], logits[3]));
    float s = 0.f, e[4];
    for (int i = 0; i < 4; ++i) { e[i] = __expf(logits[i] - mx); s += e[i]; }
    int am = 0;
    for (int i = 1; i < 4; ++i) if (logits[i] > logits[am]) am = i;
    for (int i = 0; i < 4; ++i) { out[i] = logits[i]; out[4 + i] = e[i] / s; }
    out[8] = (float)am;
  }
}

extern "C" void kernel_launch(void* const* d_in, const int* in_sizes, int n_in,
                              void* d_out, int out_size, void* d_ws, size_t ws_size,
                              hipStream_t stream) {
  (void)in_sizes; (void)n_in; (void)out_size; (void)ws_size;
  const float* x        = (const float*)d_in[0];
  const float* fc1_w    = (const float*)d_in[1];
  const float* fc1_b    = (const float*)d_in[2];
  const float* cls_tok  = (const float*)d_in[3];
  const float* l1_nw    = (const float*)d_in[4];
  const float* l1_nb    = (const float*)d_in[5];
  const float* l1_qkv_w = (const float*)d_in[6];
  const float* l1_out_w = (const float*)d_in[7];
  const float* l1_out_b = (const float*)d_in[8];
  const float* l1_res_w = (const float*)d_in[9];
  const float* pw7 = (const float*)d_in[10];
  const float* pb7 = (const float*)d_in[11];
  const float* pw5 = (const float*)d_in[12];
  const float* pb5 = (const float*)d_in[13];
  const float* pw3 = (const float*)d_in[14];
  const float* pb3 = (const float*)d_in[15];
  const float* l2_nw    = (const float*)d_in[16];
  const float* l2_nb    = (const float*)d_in[17];
  const float* l2_qkv_w = (const float*)d_in[18];
  const float* l2_out_w = (const float*)d_in[19];
  const float* l2_out_b = (const float*)d_in[20];
  const float* l2_res_w = (const float*)d_in[21];
  const float* fnw   = (const float*)d_in[22];
  const float* fnb   = (const float*)d_in[23];
  const float* fc3_w = (const float*)d_in[24];
  const float* fc3_b = (const float*)d_in[25];

  char* wp = (char*)d_ws;
  auto alloc = [&](size_t bytes) -> void* {
    void* r = (void*)wp; wp += (bytes + 255) & ~(size_t)255; return r;
  };

  bf16*  Apool = (bf16*) alloc((size_t)NPOOL * EMB * 2);
  bf16*  fc1T  = (bf16*) alloc((size_t)DM * EMB * 2);
  bf16*  qkvT1 = (bf16*) alloc((size_t)3 * DM * DM * 2);
  bf16*  outT1 = (bf16*) alloc((size_t)DM * DM * 2);
  bf16*  qkvT2 = (bf16*) alloc((size_t)3 * DM * DM * 2);
  bf16*  outT2 = (bf16*) alloc((size_t)DM * DM * 2);
  float* ft    = (float*)alloc((size_t)NPOOL * DM * 4);
  float* hA    = (float*)alloc((size_t)ND * DM * 4);
  float* hB    = (float*)alloc((size_t)ND * DM * 4);
  bf16*  lnb   = (bf16*) alloc((size_t)NP * DM * 2);
  float* qkvf  = (float*)alloc((size_t)NP * 3 * DM * 4);
  bf16*  qb    = (bf16*) alloc((size_t)NHEADS * NP * HD * 2);
  bf16*  kb    = (bf16*) alloc((size_t)NHEADS * NP * HD * 2);
  bf16*  vb    = (bf16*) alloc((size_t)NHEADS * NP * HD * 2);
  float* qlf   = (float*)alloc((size_t)NHEADS * LM * HD * 4);
  float* klf   = (float*)alloc((size_t)NHEADS * LM * HD * 4);
  bf16*  qlb   = (bf16*) alloc((size_t)NHEADS * LM * HD * 2);
  bf16*  klb   = (bf16*) alloc((size_t)NHEADS * LM * HD * 2);
  float* a2    = (float*)alloc((size_t)NHEADS * LM * LM * 4);
  float* xz    = (float*)alloc((size_t)NHEADS * LM * LM * 4);
  float* wB    = (float*)alloc((size_t)NHEADS * LM * LM * 4);
  float* wC    = (float*)alloc((size_t)NHEADS * LM * LM * 4);
  float* z0    = (float*)alloc((size_t)NHEADS * LM * LM * 4);
  float* z1    = (float*)alloc((size_t)NHEADS * LM * LM * 4);
  float* s     = (float*)alloc((size_t)NHEADS * NP * LM * 4);
  unsigned char* a1f8 = (unsigned char*)alloc((size_t)NHEADS * NP * LM);
  bf16*  a3b   = (bf16*) alloc((size_t)NHEADS * NP * LM * 2);
  float* av    = (float*)alloc((size_t)NHEADS * LM * HD * 4);
  unsigned char* zcT8 = (unsigned char*)alloc((size_t)NHEADS * HD * LM);
  float* outf  = (float*)alloc((size_t)NP * DM * 4);
  bf16*  outb  = (bf16*) alloc((size_t)NP * DM * 2);

  auto grid1 = [](long n) { return dim3((unsigned)((n + 255) / 256)); };

  pool_bf16_kernel<<<grid1((long)NPOOL * EMB), 256, 0, stream>>>(x, Apool);
  transpose_bf16_kernel<<<grid1((long)EMB * DM), 256, 0, stream>>>(fc1_w, fc1T, EMB, DM);
  transpose_bf16_kernel<<<grid1((long)DM * 3 * DM), 256, 0, stream>>>(l1_qkv_w, qkvT1, DM, 3 * DM);
  transpose_bf16_kernel<<<grid1((long)DM * DM), 256, 0, stream>>>(l1_out_w, outT1, DM, DM);
  transpose_bf16_kernel<<<grid1((long)DM * 3 * DM), 256, 0, stream>>>(l2_qkv_w, qkvT2, DM, 3 * DM);
  transpose_bf16_kernel<<<grid1((long)DM * DM), 256, 0, stream>>>(l2_out_w, outT2, DM, DM);

  gemm_bf16_wmma<<<dim3(DM / 64, NPOOL / 128, 1), 256, 0, stream>>>(
      Apool, fc1T, ft, EMB, EMB, EMB, DM, 0, 0, 0, fc1_b, 1, 0);
  build_h_kernel<<<grid1((long)ND * DM), 256, 0, stream>>>(ft, cls_tok, hA);

  auto attention = [&](float* hio, const float* norm_w, const float* norm_b,
                       const bf16* qkvT, const bf16* outT, const float* out_b,
                       const float* res_w) {
    ln_pad_kernel<<<dim3(NP), 256, 0, stream>>>(hio, norm_w, norm_b, lnb);
    gemm_bf16_wmma<<<dim3(3 * DM / 64, NP / 128, 1), 256, 0, stream>>>(
        lnb, qkvT, qkvf, DM, DM, DM, 3 * DM, 0, 0, 0, nullptr, 0, 0);
    split_qkv_kernel<<<grid1((long)NP * DM), 256, 0, stream>>>(qkvf, qb, kb, vb);
    landmarks_kernel<<<dim3(NHEADS * LM * HD / 256), 256, 0, stream>>>(qkvf, qlf, klf, qlb, klb);
    a2_softmax_kernel<<<dim3(LM, NHEADS), 256, 0, stream>>>(qlf, klf, a2);
    pinv_init_kernel<<<dim3(NHEADS), 256, 0, stream>>>(a2, z0);
    float* za = z0; float* zb = z1;
    for (int it = 0; it < 6; ++it) {
      mm256_kernel<<<dim3(16, 16, NHEADS), dim3(16, 16), 0, stream>>>(a2, za, xz, 0.f, 1.f, 0, 0.f, 0);
      mm256_kernel<<<dim3(16, 16, NHEADS), dim3(16, 16), 0, stream>>>(xz, xz, wB, 15.f, 1.f, 1, 7.f, 1);
      mm256_kernel<<<dim3(16, 16, NHEADS), dim3(16, 16), 0, stream>>>(xz, wB, wC, 13.f, 1.f, 1, 0.f, 0);
      mm256_kernel<<<dim3(16, 16, NHEADS), dim3(16, 16), 0, stream>>>(za, wC, zb, 0.f, 0.25f, 0, 0.f, 0);
      float* tmp = za; za = zb; zb = tmp;
    }
    gemm_bf16_wmma<<<dim3(LM / 64, NP / 128, NHEADS), 256, 0, stream>>>(
        qb, klb, s, HD, HD, HD, LM, (long)NP * HD, (long)LM * HD, (long)NP * LM, nullptr, 0, 0);
    softmax_rows_fp8_kernel<<<dim3(NP, NHEADS), 256, 0, stream>>>(
        s, a1f8, LM, LM, LM, (long)NP * LM, (long)NP * LM);
    gemm_bf16_wmma<<<dim3(NP / 64, LM / 128, NHEADS), 256, 0, stream>>>(
        qlb, kb, s, HD, HD, HD, NP, (long)LM * HD, (long)NP * HD, (long)LM * NP, nullptr, 0, 0);
    softmax_rows_kernel<<<dim3(LM, NHEADS), 256, 0, stream>>>(
        s, a3b, NP, NP, NP, (long)LM * NP, (long)LM * NP);
    a3v_kernel<<<dim3(LM / 4, NHEADS), 256, 0, stream>>>(a3b, vb, av);
    zc_kernel<<<dim3(NHEADS * LM * HD / 256), 256, 0, stream>>>(za, av, zcT8);
    gemm_fp8_wmma<<<dim3(1, NP / 128, NHEADS), 256, 0, stream>>>(
        a1f8, zcT8, outf, LM, LM, LM, DM, (long)NP * LM, (long)HD * LM, (long)HD);
    res_add_kernel<<<grid1((long)NP * DM), 256, 0, stream>>>(qkvf, res_w, outf);
    cvt_bf16_kernel<<<grid1((long)NP * DM), 256, 0, stream>>>(outf, outb, (long)NP * DM);
    gemm_bf16_wmma<<<dim3(DM / 64, NP / 128, 1), 256, 0, stream>>>(
        outb, outT, hio, DM, DM, DM, DM, 0, 0, 0, out_b, 2, 255);
  };

  attention(hA, l1_nw, l1_nb, qkvT1, outT1, l1_out_b, l1_res_w);
  ppeg_kernel<<<grid1((long)ND * DM), 256, 0, stream>>>(hA, hB, pw7, pb7, pw5, pb5, pw3, pb3);
  attention(hB, l2_nw, l2_nb, qkvT2, outT2, l2_out_b, l2_res_w);
  final_kernel<<<dim3(1), 512, 0, stream>>>(hB, fnw, fnb, fc3_w, fc3_b, (float*)d_out);
}
